// WuBuSparseAttention_5068061409567
// MI455X (gfx1250) — matrix operand order repaired and checked by reading, not compiled, not run on hardware
//
#include <hip/hip_runtime.h>

// ---------------------------------------------------------------------------
// Problem constants (match the reference)
// ---------------------------------------------------------------------------
constexpr int Dm   = 1024;          // model dim
constexpr int Hh   = 16;            // heads
constexpr int DH   = 64;            // head dim
constexpr int KTOP = 64;            // top-k gathered keys
constexpr int Ww   = 512;           // work window (last W positions)
constexpr int IDXD = 64;            // index-proj dim
constexpr int Ss   = 2048;          // sequence length (B = 1)
constexpr int Aa   = Ss - Ww;       // 1536 associative positions

// ---------------------------------------------------------------------------
// bf16 helpers (raw ushort storage in workspace; __bf16 for WMMA frags)
// ---------------------------------------------------------------------------
typedef __bf16 bf16_t;
typedef __attribute__((ext_vector_type(16))) __bf16 v16bf;
typedef __attribute__((ext_vector_type(8)))  __bf16 v8bf;
typedef __attribute__((ext_vector_type(8)))  float  v8f;

__device__ __forceinline__ unsigned short f2us_rn(float f) {
  union { float f; unsigned u; } v; v.f = f;
  unsigned r = v.u + 0x7FFFu + ((v.u >> 16) & 1u);   // round-to-nearest-even
  return (unsigned short)(r >> 16);
}
__device__ __forceinline__ float us2f(unsigned short h) {
  union { unsigned u; float f; } v; v.u = ((unsigned)h) << 16;
  return v.f;
}
__device__ __forceinline__ bf16_t f2bf(float f) {
  unsigned short u = f2us_rn(f);
  bf16_t b; __builtin_memcpy(&b, &u, 2); return b;
}
__device__ __forceinline__ float bf2f(bf16_t b) {
  unsigned short u; __builtin_memcpy(&u, &b, 2); return us2f(u);
}

__device__ __forceinline__ v8f wmma_bf16(v16bf a, v16bf b, v8f c) {
  // v_wmma_f32_16x16x32_bf16  D = A x B + C   (wave32, f32 accumulate)
  return __builtin_amdgcn_wmma_f32_16x16x32_bf16(
      /*neg_a=*/false, a, /*neg_b=*/false, b,
      /*c_mod=*/(short)0, c, /*reuse_a=*/false, /*reuse_b=*/false);
}

// ---------------------------------------------------------------------------
// Kernel 0: one-shot fp32 -> bf16 conversion (vectorized 4-wide)
// ---------------------------------------------------------------------------
__global__ __launch_bounds__(256)
void f32_to_bf16_kernel(const float* __restrict__ src,
                        unsigned short* __restrict__ dst, int n4)
{
  const int i = blockIdx.x * 256 + threadIdx.x;
  if (i < n4) {
    const float4 f = ((const float4*)src)[i];
    ushort4 o;
    o.x = f2us_rn(f.x); o.y = f2us_rn(f.y);
    o.z = f2us_rn(f.z); o.w = f2us_rn(f.w);
    ((ushort4*)dst)[i] = o;
  }
}

// ---------------------------------------------------------------------------
// Kernel 1: pure-bf16 WMMA GEMM   C[M,N] = A[M,Kd] * W[N,Kd]^T + bias[N]
//   Each wave computes a 16x64 strip: 1 A-fragment reused for 4 B tiles.
//   OUT_MODE: 0 = f32 [M][N], 2 = bf16 head-major [n/64][m][n%64]
// ---------------------------------------------------------------------------
template <int OUT_MODE>
__global__ __launch_bounds__(256)
void gemm_bf16_wmma_kernel(const unsigned short* __restrict__ Abf,
                           const unsigned short* __restrict__ Wbf,
                           const float* __restrict__ bias,
                           void* __restrict__ Out,
                           int M, int N, int Kd)
{
  const int lane = threadIdx.x & 31;
  const int wave = threadIdx.x >> 5;
  const int gw   = blockIdx.x * 8 + wave;       // global wave id
  const int tn4c = N >> 6;                       // 64-col tiles
  const int tm   = gw / tn4c;
  const int tn4  = gw % tn4c;
  if (tm * 16 >= M) return;                      // uniform per wave

  const int mrow = tm * 16 + (lane & 15);
  const int half = lane >> 4;

  const v8f vzero = {0.f,0.f,0.f,0.f,0.f,0.f,0.f,0.f};
  v8f acc[4];
#pragma unroll
  for (int j = 0; j < 4; ++j) acc[j] = vzero;

  const unsigned short* arow = Abf + (size_t)mrow * Kd;

#pragma unroll 2
  for (int k0 = 0; k0 < Kd; k0 += 32) {
    // ---- A fragment: 16x32 bf16, two aligned 16B loads (ISA 7.12.2) ------
    union { v16bf v; v8bf h[2]; } ua;
    ua.h[0] = *(const v8bf*)(arow + k0 + half * 16);
    ua.h[1] = *(const v8bf*)(arow + k0 + half * 16 + 8);
    const v16bf afrag = ua.v;
    __builtin_prefetch((const void*)(arow + k0 + 64), 0, 1);   // next chunk

    // ---- 4 B fragments (32x16 bf16): one aligned 32B load each -----------
#pragma unroll
    for (int j = 0; j < 4; ++j) {
      const int ncol = tn4 * 64 + j * 16 + (lane & 15);
      const v16bf bfrag =
          *(const v16bf*)(Wbf + (size_t)ncol * Kd + k0 + half * 16);
      acc[j] = wmma_bf16(afrag, bfrag, acc[j]);
    }
  }

  // ---- bias + store (C/D layout: lane n = lane&15; row m = r + 8*half) ---
#pragma unroll
  for (int j = 0; j < 4; ++j) {
    const int ncol = tn4 * 64 + j * 16 + (lane & 15);
    const float bv = bias[ncol];
#pragma unroll
    for (int r = 0; r < 8; ++r) {
      const int m = tm * 16 + r + half * 8;
      const float val = acc[j][r] + bv;
      if constexpr (OUT_MODE == 0) {
        ((float*)Out)[(size_t)m * N + ncol] = val;
      } else {
        const int h = ncol >> 6, d = ncol & 63;
        ((unsigned short*)Out)[((size_t)h * M + m) * DH + d] = f2us_rn(val);
      }
    }
  }
}

// ---------------------------------------------------------------------------
// Kernel 2: index scores + iterative top-64 (one block per query row)
// ---------------------------------------------------------------------------
__global__ __launch_bounds__(256)
void topk_kernel(const float* __restrict__ qidx,   // [S][IDXD]
                 const float* __restrict__ kidx,   // [A][IDXD]
                 int* __restrict__ topidx)         // [S][KTOP]
{
  __shared__ float sc[Aa];
  __shared__ float qrow[IDXD];
  __shared__ float red_v[256];
  __shared__ int   red_i[256];

  const int s   = blockIdx.x;
  const int tid = threadIdx.x;

  if (tid < IDXD) qrow[tid] = qidx[(size_t)s * IDXD + tid];
  __syncthreads();

  for (int a = tid; a < Aa; a += 256) {
    const float* kr = kidx + (size_t)a * IDXD;
    float acc = 0.f;
#pragma unroll 8
    for (int e = 0; e < IDXD; ++e) acc += qrow[e] * kr[e];
    sc[a] = acc > 0.f ? acc : 0.f;                 // relu
  }
  __syncthreads();

  for (int it = 0; it < KTOP; ++it) {
    float bv = -1e30f; int bi = 0x7fffffff;
    for (int a = tid; a < Aa; a += 256) {
      const float v = sc[a];
      if (v > bv || (v == bv && a < bi)) { bv = v; bi = a; }
    }
    red_v[tid] = bv; red_i[tid] = bi;
    __syncthreads();
    for (int off = 128; off > 0; off >>= 1) {
      if (tid < off) {
        const float v2 = red_v[tid + off]; const int i2 = red_i[tid + off];
        if (v2 > red_v[tid] || (v2 == red_v[tid] && i2 < red_i[tid])) {
          red_v[tid] = v2; red_i[tid] = i2;
        }
      }
      __syncthreads();
    }
    const int win = red_i[0];
    __syncthreads();
    if (tid == 0) { topidx[(size_t)s * KTOP + it] = win; sc[win] = -1e30f; }
    __syncthreads();
  }
}

// ---------------------------------------------------------------------------
// Kernel 3: attention — one block (8 waves) per (head, 16-query tile)
//   All WMMA fragments are direct aligned vector loads (global or LDS bf16).
// ---------------------------------------------------------------------------
__global__ __launch_bounds__(256)
void attn_kernel(const unsigned short* __restrict__ q_ws,  // [H][S][DH] bf16
                 const unsigned short* __restrict__ k_ws,
                 const unsigned short* __restrict__ v_ws,
                 const int* __restrict__ topidx,            // [S][KTOP]
                 unsigned short* __restrict__ attn_out)     // [S][D] bf16
{
  __shared__ __align__(32) bf16_t qsb[16][72];          // Q tile, bf16
  __shared__ float sc[16][KTOP + Ww + 1];               // 576 scores, f32
  __shared__ float oacc[16][DH + 1];
  __shared__ int   sidx[16][KTOP];
  __shared__ __align__(32) bf16_t wwb[16][520];         // work weights, bf16
  __shared__ __align__(32) bf16_t vtb[DH][528];         // V-work transposed

  const int h    = blockIdx.x >> 7;             // 128 query tiles per head
  const int qt   = blockIdx.x & 127;
  const int s0   = qt * 16;
  const int tid  = threadIdx.x;
  const int lane = tid & 31;
  const int wave = tid >> 5;
  const float scale = 0.125f;                   // 1/sqrt(64)

  // ---- stage Q tile (bf16), top-k indices, and V-work transposed ---------
  for (int i = tid; i < 16 * DH / 8; i += 256) {        // 128 vec8 chunks
    const int q = i >> 3, db = (i & 7) * 8;
    *(v8bf*)&qsb[q][db] =
        *(const v8bf*)(q_ws + ((size_t)h * Ss + s0 + q) * DH + db);
  }
  for (int i = tid; i < 16 * KTOP; i += 256)
    sidx[i >> 6][i & 63] = topidx[(size_t)(s0 + (i >> 6)) * KTOP + (i & 63)];
  for (int i = tid; i < (Ww * DH) / 8; i += 256) {      // 4096 vec8 chunks
    const int key = i >> 3, db = (i & 7) * 8;
    const v8bf vv =
        *(const v8bf*)(v_ws + ((size_t)h * Ss + Aa + key) * DH + db);
#pragma unroll
    for (int e = 0; e < 8; ++e) vtb[db + e][key] = vv[e];
  }
  __syncthreads();

  // ---- work scores: Q(16x64) x Kwork^T(64x512), 4 n-tiles per wave -------
  for (int t = wave * 4; t < wave * 4 + 4; ++t) {
    v8f acc = {0.f,0.f,0.f,0.f,0.f,0.f,0.f,0.f};
#pragma unroll
    for (int ks = 0; ks < 2; ++ks) {
      const int k0 = ks * 32;
      const int m = lane & 15, hf = lane >> 4;
      union { v16bf v; v8bf h[2]; } ua;
      ua.h[0] = *(const v8bf*)&qsb[m][k0 + hf * 16];
      ua.h[1] = *(const v8bf*)&qsb[m][k0 + hf * 16 + 8];
      const int key = t * 16 + m;
      const v16bf bfrag = *(const v16bf*)(
          k_ws + ((size_t)h * Ss + Aa + key) * DH + k0 + hf * 16);
      acc = wmma_bf16(ua.v, bfrag, acc);
    }
    const int n = lane & 15, hf = lane >> 4;
#pragma unroll
    for (int r = 0; r < 8; ++r) sc[r + hf * 8][KTOP + t * 16 + n] = acc[r] * scale;
  }
  __syncthreads();

  // ---- gathered (associative) scores: 16 q x 64 keys, vec8 dot -----------
  for (int i = tid; i < 16 * KTOP; i += 256) {
    const int q = i >> 6, j = i & 63;
    const unsigned short* kp = k_ws + ((size_t)h * Ss + sidx[q][j]) * DH;
    float a = 0.f;
#pragma unroll
    for (int c = 0; c < 8; ++c) {
      const v8bf kv = *(const v8bf*)(kp + c * 8);
      const v8bf qv = *(const v8bf*)&qsb[q][c * 8];
#pragma unroll
      for (int e = 0; e < 8; ++e) a += bf2f(qv[e]) * bf2f(kv[e]);
    }
    sc[q][j] = a * scale;
  }
  __syncthreads();

  // ---- softmax over 576 per row: 2 rows per wave, shfl reductions --------
  for (int q = wave * 2; q < wave * 2 + 2; ++q) {
    float mx = -1e30f;
    for (int j = lane; j < KTOP + Ww; j += 32) mx = fmaxf(mx, sc[q][j]);
#pragma unroll
    for (int o = 16; o > 0; o >>= 1) mx = fmaxf(mx, __shfl_xor(mx, o, 32));
    float sum = 0.f;
    for (int j = lane; j < KTOP + Ww; j += 32) {
      const float e = __expf(sc[q][j] - mx);
      sc[q][j] = e; sum += e;
    }
#pragma unroll
    for (int o = 16; o > 0; o >>= 1) sum += __shfl_xor(sum, o, 32);
    const float inv = 1.f / sum;
    for (int j = lane; j < KTOP + Ww; j += 32) sc[q][j] *= inv;
  }
  __syncthreads();

  // ---- bf16 copy of work weights + gathered V contribution ---------------
  for (int i = tid; i < 16 * Ww; i += 256) {            // weights -> bf16 LDS
    const int q = i >> 9, k = i & 511;
    wwb[q][k] = f2bf(sc[q][KTOP + k]);
  }
  for (int i = tid; i < 16 * 8; i += 256) {             // (q, d-block of 8)
    const int q = i >> 3, db = (i & 7) * 8;
    float a[8] = {0.f,0.f,0.f,0.f,0.f,0.f,0.f,0.f};
    for (int j = 0; j < KTOP; ++j) {
      const v8bf vv =
          *(const v8bf*)(v_ws + ((size_t)h * Ss + sidx[q][j]) * DH + db);
      const float w = sc[q][j];
#pragma unroll
      for (int e = 0; e < 8; ++e) a[e] += w * bf2f(vv[e]);
    }
#pragma unroll
    for (int e = 0; e < 8; ++e) oacc[q][db + e] = a[e];
  }
  __syncthreads();

  // ---- work V contribution: weights(16x512) x Vwork(512x64) via WMMA -----
  {
    const int nt = wave & 3;          // d-tile (4 tiles of 16)
    const int kh = wave >> 2;         // split K=512 across wave pairs
    v8f acc = {0.f,0.f,0.f,0.f,0.f,0.f,0.f,0.f};
#pragma unroll
    for (int ks = 0; ks < 8; ++ks) {
      const int k0 = kh * 256 + ks * 32;
      const int m = lane & 15, hf = lane >> 4;
      union { v16bf v; v8bf h[2]; } ua;
      ua.h[0] = *(const v8bf*)&wwb[m][k0 + hf * 16];
      ua.h[1] = *(const v8bf*)&wwb[m][k0 + hf * 16 + 8];
      const int d  = nt * 16 + m;
      const v16bf bfrag = *(const v16bf*)&vtb[d][k0 + hf * 16];
      acc = wmma_bf16(ua.v, bfrag, acc);
    }
    const int n = nt * 16 + (lane & 15), hf = lane >> 4;
#pragma unroll
    for (int r = 0; r < 8; ++r)
      atomicAdd(&oacc[r + hf * 8][n], acc[r]);   // ds_add_f32; 2 waves/region
  }
  __syncthreads();

  // ---- write bf16 attention output [s][h*64+d] ---------------------------
  for (int i = tid; i < 16 * DH; i += 256) {
    const int q = i >> 6, d = i & 63;
    attn_out[(size_t)(s0 + q) * Dm + h * DH + d] = f2us_rn(oacc[q][d]);
  }
}

// ---------------------------------------------------------------------------
// Host launch
// ---------------------------------------------------------------------------
extern "C" void kernel_launch(void* const* d_in, const int* in_sizes, int n_in,
                              void* d_out, int out_size, void* d_ws, size_t ws_size,
                              hipStream_t stream) {
  const float* x   = (const float*)d_in[0];
  const float* wq  = (const float*)d_in[1];
  const float* bq  = (const float*)d_in[2];
  const float* wk  = (const float*)d_in[3];
  const float* bk  = (const float*)d_in[4];
  const float* wv  = (const float*)d_in[5];
  const float* bv  = (const float*)d_in[6];
  const float* wo  = (const float*)d_in[7];
  const float* bo  = (const float*)d_in[8];
  const float* wqi = (const float*)d_in[9];
  const float* bqi = (const float*)d_in[10];
  const float* wki = (const float*)d_in[11];
  const float* bki = (const float*)d_in[12];

  char* ws = (char*)d_ws;
  size_t off = 0;
  auto carve = [&](size_t bytes) {
    void* p = ws + off;
    off = (off + bytes + 255) & ~(size_t)255;
    return p;
  };
  // bf16 copies of activations + weights (converted once)
  unsigned short* x_bf    = (unsigned short*)carve((size_t)Ss * Dm * 2);
  unsigned short* wq_bf   = (unsigned short*)carve((size_t)Dm * Dm * 2);
  unsigned short* wk_bf   = (unsigned short*)carve((size_t)Dm * Dm * 2);
  unsigned short* wv_bf   = (unsigned short*)carve((size_t)Dm * Dm * 2);
  unsigned short* wo_bf   = (unsigned short*)carve((size_t)Dm * Dm * 2);
  unsigned short* wqi_bf  = (unsigned short*)carve((size_t)IDXD * Dm * 2);
  unsigned short* wki_bf  = (unsigned short*)carve((size_t)IDXD * Dm * 2);
  // intermediates
  unsigned short* q_ws    = (unsigned short*)carve((size_t)Ss * Dm * 2);  // bf16 head-major
  unsigned short* k_ws    = (unsigned short*)carve((size_t)Ss * Dm * 2);
  unsigned short* v_ws    = (unsigned short*)carve((size_t)Ss * Dm * 2);
  float*          qidx_f  = (float*)carve((size_t)Ss * IDXD * 4);
  float*          kidx_f  = (float*)carve((size_t)Aa * IDXD * 4);
  int*            topidx  = (int*)carve((size_t)Ss * KTOP * 4);
  unsigned short* attn_ws = (unsigned short*)carve((size_t)Ss * Dm * 2);
  (void)ws_size; (void)in_sizes; (void)n_in; (void)out_size;

  const dim3 blk(256);

  // ---- one-shot fp32 -> bf16 conversions ---------------------------------
  auto conv = [&](const float* s, unsigned short* d, int n) {
    const int n4 = n / 4;
    f32_to_bf16_kernel<<<dim3((n4 + 255) / 256), blk, 0, stream>>>(s, d, n4);
  };
  conv(x,   x_bf,   Ss * Dm);
  conv(wq,  wq_bf,  Dm * Dm);
  conv(wk,  wk_bf,  Dm * Dm);
  conv(wv,  wv_bf,  Dm * Dm);
  conv(wo,  wo_bf,  Dm * Dm);
  conv(wqi, wqi_bf, IDXD * Dm);
  conv(wki, wki_bf, IDXD * Dm);

  // ---- Q/K/V projections: (2048/16)*(1024/64) = 2048 waves -> 256 blocks -
  gemm_bf16_wmma_kernel<2><<<dim3(256), blk, 0, stream>>>(x_bf, wq_bf, bq, q_ws, Ss, Dm, Dm);
  gemm_bf16_wmma_kernel<2><<<dim3(256), blk, 0, stream>>>(x_bf, wk_bf, bk, k_ws, Ss, Dm, Dm);
  gemm_bf16_wmma_kernel<2><<<dim3(256), blk, 0, stream>>>(x_bf, wv_bf, bv, v_ws, Ss, Dm, Dm);
  // index projections: N=64 -> (M/16)*1 waves
  gemm_bf16_wmma_kernel<0><<<dim3(16), blk, 0, stream>>>(x_bf, wqi_bf, bqi, qidx_f, Ss, IDXD, Dm);
  gemm_bf16_wmma_kernel<0><<<dim3(12), blk, 0, stream>>>(x_bf, wki_bf, bki, kidx_f, Aa, IDXD, Dm);
  // top-64 selection per query
  topk_kernel<<<dim3(Ss), blk, 0, stream>>>(qidx_f, kidx_f, topidx);
  // attention: H * (S/16) = 2048 blocks
  attn_kernel<<<dim3(Hh * (Ss / 16)), blk, 0, stream>>>(q_ws, k_ws, v_ws, topidx, attn_ws);
  // output projection -> fp32 d_out
  gemm_bf16_wmma_kernel<0><<<dim3(256), blk, 0, stream>>>(attn_ws, wo_bf, bo, (float*)d_out, Ss, Dm, Dm);
}
